// DecoderModel_21105469292905
// MI455X (gfx1250) — matrix operand order repaired and checked by reading, not compiled, not run on hardware
//
#include <hip/hip_runtime.h>
#include <hip/hip_bf16.h>
#include <math.h>
#include <stdint.h>

// Model dims (compile-time constants from the reference)
#define BB   4
#define TT   1024
#define DD   1024
#define HH   16
#define LL   12
#define VV   50257
#define DHH  64
#define BT_  (BB*TT)          // 4096 rows
#define D4   (4*DD)           // 4096 MLP hidden

typedef __bf16 bf16;
typedef __attribute__((ext_vector_type(16))) __bf16 v16bf;
typedef __attribute__((ext_vector_type(8)))  __bf16 v8bf;
typedef __attribute__((ext_vector_type(4)))  __bf16 v4bf;
typedef __attribute__((ext_vector_type(8)))  float  v8f;
typedef __attribute__((ext_vector_type(4)))  unsigned int u32x4;
typedef __attribute__((ext_vector_type(8)))  int  i32x8;
typedef __attribute__((ext_vector_type(4)))  int  i32x4;

static __device__ __forceinline__ v16bf cat8(v8bf lo, v8bf hi) {
  v16bf r;
#pragma unroll
  for (int i = 0; i < 8; ++i) { r[i] = lo[i]; r[i + 8] = hi[i]; }
  return r;
}

static __device__ __forceinline__ v8f wmma_bf16(v16bf a, v16bf b, v8f c) {
  return __builtin_amdgcn_wmma_f32_16x16x32_bf16(
      /*neg_a=*/false, a, /*neg_b=*/false, b,
      /*c_mod=*/(short)0, c, /*reuse_a=*/false, /*reuse_b=*/false);
}

// ---------------------------------------------------------------------------
// Embedding: x[b,t,:] = tok_emb[idx[b,t],:] + pos_emb[t,:]
// ---------------------------------------------------------------------------
__global__ void k_embed(const int* __restrict__ idx,
                        const float* __restrict__ tok,
                        const float* __restrict__ pos,
                        float* __restrict__ x) {
  int row = blockIdx.x;              // 0..BT-1
  int t   = row % TT;
  int tk  = idx[row];
  const float* te = tok + (size_t)tk * DD;
  const float* pe = pos + (size_t)t * DD;
  float* xr = x + (size_t)row * DD;
  for (int c = threadIdx.x; c < DD; c += blockDim.x) xr[c] = te[c] + pe[c];
}

// ---------------------------------------------------------------------------
// LayerNorm: fp32 in -> bf16 out (normalized*scale+bias), one block per row
// ---------------------------------------------------------------------------
__global__ void k_ln(const float* __restrict__ x,
                     const float* __restrict__ s,
                     const float* __restrict__ b,
                     bf16* __restrict__ out) {
  int row = blockIdx.x;
  const float* xr = x + (size_t)row * DD;
  int base = threadIdx.x * 4;        // 256 threads * 4 = 1024
  float v0[4]; float sum = 0.f, sq = 0.f;
#pragma unroll
  for (int i = 0; i < 4; ++i) { float f = xr[base + i]; v0[i] = f; sum += f; sq += f * f; }
  for (int m = 16; m >= 1; m >>= 1) { sum += __shfl_xor(sum, m, 32); sq += __shfl_xor(sq, m, 32); }
  __shared__ float ssum[8], ssq[8];
  int w = threadIdx.x >> 5, ln = threadIdx.x & 31;
  if (ln == 0) { ssum[w] = sum; ssq[w] = sq; }
  __syncthreads();
  if (w == 0) {
    float a = (ln < 8) ? ssum[ln] : 0.f;
    float c = (ln < 8) ? ssq[ln] : 0.f;
    for (int m = 4; m >= 1; m >>= 1) { a += __shfl_xor(a, m, 32); c += __shfl_xor(c, m, 32); }
    if (ln == 0) { ssum[0] = a; ssq[0] = c; }
  }
  __syncthreads();
  float mu   = ssum[0] * (1.f / DD);
  float var  = ssq[0] * (1.f / DD) - mu * mu;
  float rstd = rsqrtf(var + 1e-5f);
  bf16* orow = out + (size_t)row * DD;
#pragma unroll
  for (int i = 0; i < 4; ++i) {
    int c = base + i;
    orow[c] = (bf16)(((v0[i] - mu) * rstd) * s[c] + b[c]);
  }
}

// ---------------------------------------------------------------------------
// WMMA GEMM: C[M,N] = A[M,K](bf16) x Bw[K,N](fp32, converted to bf16 in LDS)
//   optional: + bias[N], ReLU, + resid[M,N](fp32). Writes outF and/or outH.
// Workgroup: 128 threads (4 waves), 128x64 tile; each wave -> 32x64 of C
// (2 A-fragments x 4 B-fragments = 8 WMMAs per 32-deep K step).
// A tile is staged by the Tensor Data Mover (tensor_load_to_lds, TENSORcnt);
// B tile is staged with packed f32->bf16 conversion and b64 LDS stores.
// Edge column blocks use clamp+select (branch-free) instead of exec masking.
// Requires: M % 128 == 0, K % 32 == 0 (true for all call sites).
// ---------------------------------------------------------------------------
__global__ void k_gemm(const bf16* __restrict__ A, const float* __restrict__ Bw,
                       const float* __restrict__ bias, const float* __restrict__ resid,
                       float* __restrict__ outF, bf16* __restrict__ outH,
                       int M, int N, int K, int relu) {
  __shared__ bf16 as[128 * 32];  // A tile, [row][k]  (written by TDM)
  __shared__ bf16 bs[64 * 32];   // B tile, [n][k] (transposed -> contiguous frags)
  const int tid   = threadIdx.x;           // 0..127
  const int lane  = tid & 31;
  const int w     = tid >> 5;
  const int row0  = blockIdx.y * 128;
  const int col0  = blockIdx.x * 64;
  const int hr    = lane & 15;
  const int kb    = (lane < 16) ? 0 : 16;  // A-fragment contiguous K halves
  const int koffB = (lane < 16) ? 0 : 16;  // B-fragment contiguous K halves
  const int mo    = (lane < 16) ? 0 : 8;   // C row offset per lane half
  const bool fullN = (col0 + 64 <= N);     // uniform: interior column block?

  // --- Tensor DMA descriptor (D#) for the A tile: 2D tile 32(k) x 128(rows),
  //     element 2B, tensor_dim0_stride = K. Groups per ISA 08_async_tensor §8.
  const unsigned ldsA = (unsigned)(uintptr_t)&as[0];
  i32x8 g1;
#pragma unroll
  for (int i = 0; i < 8; ++i) g1[i] = 0;
  g1[0] = (1 << 16);                                    // data_size = 1 (2 bytes)
  g1[1] = (int)((K & 0xffffu) << 16);                   // tensor_dim0[15:0]
  g1[2] = (int)(((unsigned)K >> 16) | ((M & 0xffffu) << 16)); // dim0 hi | dim1 lo
  g1[3] = (int)(((unsigned)M >> 16) | (32u << 16));     // dim1 hi | tile_dim0 = 32
  g1[4] = 128;                                          // tile_dim1 = 128, tile_dim2 = 0
  g1[5] = (int)(unsigned)K;                             // tensor_dim0_stride lo32
  g1[6] = 0;                                            // stride hi | dim1_stride lo
  g1[7] = 0;
  i32x4 gz;
#pragma unroll
  for (int i = 0; i < 4; ++i) gz[i] = 0;
  i32x8 gz8;
#pragma unroll
  for (int i = 0; i < 8; ++i) gz8[i] = 0;

  v8f acc[2][4];
#pragma unroll
  for (int s = 0; s < 2; ++s)
#pragma unroll
    for (int c = 0; c < 4; ++c)
#pragma unroll
      for (int i = 0; i < 8; ++i) acc[s][c][i] = 0.f;

  const int nst   = tid & 63;                 // B-stage column (lane-coalesced)
  const int kqb   = tid >> 6;                 // B-stage K-quad base
  const int gcRaw = col0 + nst;
  const int gcCl  = (gcRaw < N) ? gcRaw : (N - 1);  // clamped, always valid
  const float nMask = (gcRaw < N) ? 1.f : 0.f;      // zero out-of-range columns

  for (int k0 = 0; k0 < K; k0 += 32) {
    // --- A tile via Tensor Data Mover (one wave issues; EXEC-independent op)
    if (w == 0) {
      uint64_t ga = (uint64_t)(uintptr_t)(A + (size_t)row0 * K + k0);
      u32x4 g0;
      g0[0] = 1u;                                       // count=1, user descriptor
      g0[1] = ldsA;                                     // LDS byte address
      g0[2] = (unsigned)ga;                             // global_addr[31:0]
      g0[3] = (unsigned)((ga >> 32) & 0x01ffffffu) | (2u << 30); // addr[56:32] | type=2
      __builtin_amdgcn_tensor_load_to_lds(g0, g1, gz, gz, gz8, 0);
    }
    // --- B tile: each thread owns column n, loads 4 consecutive-K floats per
    //     quad (lane-coalesced along n), packs to bf16, one b64 LDS store.
    //     Branch-free edge handling: clamped address + multiplicative mask.
#pragma unroll
    for (int it = 0; it < 4; ++it) {
      int kq = kqb + it * 2;
      const float* src = &Bw[(size_t)(k0 + kq * 4) * N + gcCl];
      v4bf pk;
#pragma unroll
      for (int j = 0; j < 4; ++j) pk[j] = (bf16)(src[(size_t)j * N] * nMask);
      *(v4bf*)&bs[nst * 32 + kq * 4] = pk;
    }
    // Prefetch next B tile into the cache hierarchy (global_prefetch_b8)
    if (k0 + 32 < K && fullN)
      __builtin_prefetch(&Bw[(size_t)(k0 + 32 + (tid >> 4)) * N + col0 + (tid & 15) * 4], 0, 1);
    if (w == 0) __builtin_amdgcn_s_wait_tensorcnt(0);   // TDM done before barrier
    __syncthreads();

    v16bf af[2];
#pragma unroll
    for (int s = 0; s < 2; ++s) {
      int ar = w * 32 + s * 16 + hr;
      af[s] = cat8(*(const v8bf*)&as[ar * 32 + kb],
                   *(const v8bf*)&as[ar * 32 + kb + 8]);
    }
#pragma unroll
    for (int c = 0; c < 4; ++c) {
      v16bf bfr = *(const v16bf*)&bs[(c * 16 + hr) * 32 + koffB];
      acc[0][c] = wmma_bf16(af[0], bfr, acc[0][c]);
      acc[1][c] = wmma_bf16(af[1], bfr, acc[1][c]);
    }
    __syncthreads();
  }

  // Epilogue: bias / relu / residual, store fp32 and/or bf16
#pragma unroll
  for (int c = 0; c < 4; ++c) {
    int col = col0 + c * 16 + hr;
    if (!fullN && col >= N) continue;
    float bi = bias ? bias[col] : 0.f;
#pragma unroll
    for (int s = 0; s < 2; ++s) {
#pragma unroll
      for (int v = 0; v < 8; ++v) {
        int row = row0 + w * 32 + s * 16 + v + mo;
        float val = acc[s][c][v] + bi;
        if (relu) val = fmaxf(val, 0.f);
        if (resid) val += resid[(size_t)row * N + col];
        if (outF) outF[(size_t)row * N + col] = val;
        if (outH) outH[(size_t)row * N + col] = (bf16)val;
      }
    }
  }
}

// ---------------------------------------------------------------------------
// Flash attention (causal), WMMA bf16. One workgroup = 64 query rows of one
// (b,h); each of 4 waves owns 16 query rows. Key blocks of 32.
// K tile staged with global_load_async_to_lds_b128 (ASYNCcnt path).
// Q,K,V,O: bf16 [B*T, D], head h occupies columns [h*64, h*64+64).
// scale = 1/sqrt(D) = 1/32 (reference divides by sqrt(d_model)).
// ---------------------------------------------------------------------------
__global__ void k_attn(const bf16* __restrict__ Q, const bf16* __restrict__ Kt,
                       const bf16* __restrict__ Vt, bf16* __restrict__ O) {
  __shared__ bf16 ks[32 * 64];        // [key][feature]
  __shared__ bf16 vt[64 * 32];        // [feature][key]  (transposed V tile)
  __shared__ bf16 pl[4 * 16 * 32];    // per-wave P transpose buffer

  const int tid   = threadIdx.x, lane = tid & 31, w = tid >> 5;
  const int q0    = blockIdx.x * 64;
  const int h     = blockIdx.y;
  const int b     = blockIdx.z;
  const int hcol  = h * DHH;
  const size_t rb = (size_t)b * TT;
  const int hr    = lane & 15;
  const int kb    = (lane < 16) ? 0 : 16;
  const int koffB = (lane < 16) ? 0 : 16;
  const int mo    = (lane < 16) ? 0 : 8;
  const int qw0   = q0 + w * 16;
  const float scale = 0.03125f;       // 1/sqrt(1024)

  // Q fragments (A-matrix layout), kept in registers: features 0..31, 32..63
  const bf16* qrow = Q + (rb + qw0 + hr) * DD + hcol;
  v16bf qa0 = cat8(*(const v8bf*)&qrow[kb],      *(const v8bf*)&qrow[kb + 8]);
  v16bf qa1 = cat8(*(const v8bf*)&qrow[32 + kb], *(const v8bf*)&qrow[32 + kb + 8]);

  v8f o[4];
  float mi[8], li[8];
#pragma unroll
  for (int c = 0; c < 4; ++c)
#pragma unroll
    for (int i = 0; i < 8; ++i) o[c][i] = 0.f;
#pragma unroll
  for (int v = 0; v < 8; ++v) { mi[v] = -3.0e38f; li[v] = 0.f; }

  bf16* pw = &pl[w * 512];
  const int nblk = q0 / 32 + 2;       // causal: keys up to q0+63

  for (int jb = 0; jb < nblk; ++jb) {
    const int j0 = jb * 32;

    // --- Stage K tile [32 keys][64 feats] via async global->LDS (32B/thread)
    {
      int base = tid * 16;            // 128 threads * 16 bf16 = 2048
      int kk = base >> 6, f = base & 63;
      const bf16* gK = &Kt[(rb + j0 + kk) * DD + hcol + f];
      unsigned ldsK = (unsigned)(uintptr_t)&ks[base];
      asm volatile(
          "global_load_async_to_lds_b128 %0, %2, off\n\t"
          "global_load_async_to_lds_b128 %1, %3, off"
          :: "v"(ldsK), "v"(ldsK + 16u), "v"(gK), "v"(gK + 8)
          : "memory");
    }
    // --- Stage V tile transposed [64 feats][32 keys]: one 32B load + scatter
    {
      int vk = tid >> 2;              // key 0..31
      int f0 = (tid & 3) * 16;        // feature group
      v16bf vv = *(const v16bf*)&Vt[(rb + j0 + vk) * DD + hcol + f0];
#pragma unroll
      for (int j = 0; j < 16; ++j) vt[(f0 + j) * 32 + vk] = vv[j];
    }
    asm volatile("s_wait_asynccnt 0" ::: "memory");
    __syncthreads();

    // Wave-uniform causal skip: this wave's rows end at qw0+15
    if (j0 <= qw0 + 15) {
      // S = Q x K^T : two 16x16 key tiles, K-depth 64 = 2 WMMAs each
      v8f s0, s1;
#pragma unroll
      for (int i = 0; i < 8; ++i) { s0[i] = 0.f; s1[i] = 0.f; }
      {
        int n0 = hr, n1 = 16 + hr;
        v16bf kb00 = *(const v16bf*)&ks[n0 * 64 + koffB];
        v16bf kb01 = *(const v16bf*)&ks[n0 * 64 + 32 + koffB];
        v16bf kb10 = *(const v16bf*)&ks[n1 * 64 + koffB];
        v16bf kb11 = *(const v16bf*)&ks[n1 * 64 + 32 + koffB];
        s0 = wmma_bf16(qa0, kb00, s0);
        s0 = wmma_bf16(qa1, kb01, s0);
        s1 = wmma_bf16(qa0, kb10, s1);
        s1 = wmma_bf16(qa1, kb11, s1);
      }

      // Scale + causal mask + online softmax (C layout: row = v+mo, col = hr)
      float rm[8];
#pragma unroll
      for (int v = 0; v < 8; ++v) {
        int qr = qw0 + v + mo;
        float a = s0[v] * scale; if (j0 + hr      > qr) a = -1.0e30f;
        float c = s1[v] * scale; if (j0 + 16 + hr > qr) c = -1.0e30f;
        s0[v] = a; s1[v] = c;
        rm[v] = fmaxf(a, c);
      }
#pragma unroll
      for (int v = 0; v < 8; ++v)
        for (int m = 8; m >= 1; m >>= 1) rm[v] = fmaxf(rm[v], __shfl_xor(rm[v], m, 32));

      float al[8];
#pragma unroll
      for (int v = 0; v < 8; ++v) {
        float mn = fmaxf(mi[v], rm[v]);
        al[v] = __expf(mi[v] - mn);
        mi[v] = mn;
        s0[v] = __expf(s0[v] - mn);
        s1[v] = __expf(s1[v] - mn);
      }
#pragma unroll
      for (int v = 0; v < 8; ++v) {
        float rs = s0[v] + s1[v];
        for (int m = 8; m >= 1; m >>= 1) rs += __shfl_xor(rs, m, 32);
        li[v] = li[v] * al[v] + rs;
#pragma unroll
        for (int c = 0; c < 4; ++c) o[c][v] *= al[v];
      }

      // Transpose P through per-wave LDS (C layout -> A-fragment layout)
#pragma unroll
      for (int v = 0; v < 8; ++v) {
        int m = v + mo;
        pw[m * 32 + hr]      = (bf16)s0[v];
        pw[m * 32 + 16 + hr] = (bf16)s1[v];
      }
      asm volatile("s_wait_dscnt 0" ::: "memory");   // same-wave LDS RAW
      v16bf pa = cat8(*(const v8bf*)&pw[hr * 32 + kb],
                      *(const v8bf*)&pw[hr * 32 + kb + 8]);

      // O += P x V : 4 feature tiles of 16
#pragma unroll
      for (int c = 0; c < 4; ++c) {
        int n = c * 16 + hr;
        v16bf vb = *(const v16bf*)&vt[n * 32 + koffB];
        o[c] = wmma_bf16(pa, vb, o[c]);
      }
    }
    __syncthreads();
  }

  // Normalize and store O (bf16), column layout matches head-concat order
#pragma unroll
  for (int c = 0; c < 4; ++c) {
#pragma unroll
    for (int v = 0; v < 8; ++v) {
      int m = v + mo;
      float val = o[c][v] / li[v];
      O[(rb + qw0 + m) * DD + hcol + c * 16 + hr] = (bf16)val;
    }
  }
}

// ---------------------------------------------------------------------------
// Host orchestration
// ---------------------------------------------------------------------------
extern "C" void kernel_launch(void* const* d_in, const int* in_sizes, int n_in,
                              void* d_out, int out_size, void* d_ws, size_t ws_size,
                              hipStream_t stream) {
  const int*   idx    = (const int*)d_in[0];
  const float* tok    = (const float*)d_in[1];
  const float* pos    = (const float*)d_in[2];
  const float* Wq     = (const float*)d_in[3];
  const float* Wk     = (const float*)d_in[4];
  const float* Wv     = (const float*)d_in[5];
  const float* Wo     = (const float*)d_in[6];
  const float* bo     = (const float*)d_in[7];
  const float* ln1_s  = (const float*)d_in[8];
  const float* ln1_b  = (const float*)d_in[9];
  const float* ln2_s  = (const float*)d_in[10];
  const float* ln2_b  = (const float*)d_in[11];
  const float* W1     = (const float*)d_in[12];
  const float* b1     = (const float*)d_in[13];
  const float* W2     = (const float*)d_in[14];
  const float* b2     = (const float*)d_in[15];
  const float* lnf_s  = (const float*)d_in[16];
  const float* lnf_b  = (const float*)d_in[17];
  const float* lm_W   = (const float*)d_in[18];
  const float* lm_b   = (const float*)d_in[19];
  float* logits = (float*)d_out;

  // Carve workspace
  char* p = (char*)d_ws;
  auto carve = [&](size_t bytes) -> void* {
    void* r = (void*)p;
    p += (bytes + 255) & ~(size_t)255;
    return r;
  };
  float* xA   = (float*)carve((size_t)BT_ * DD * sizeof(float));   // residual stream A
  float* xB   = (float*)carve((size_t)BT_ * DD * sizeof(float));   // residual stream B
  bf16*  xn   = (bf16*) carve((size_t)BT_ * DD * sizeof(bf16));    // LN output
  bf16*  q    = (bf16*) carve((size_t)BT_ * DD * sizeof(bf16));
  bf16*  k    = (bf16*) carve((size_t)BT_ * DD * sizeof(bf16));
  bf16*  v    = (bf16*) carve((size_t)BT_ * DD * sizeof(bf16));
  bf16*  attn = (bf16*) carve((size_t)BT_ * DD * sizeof(bf16));
  bf16*  hbuf = (bf16*) carve((size_t)BT_ * D4 * sizeof(bf16));

  dim3 gD((DD + 63) / 64, BT_ / 128);    // [BT x D] GEMMs
  dim3 g4((D4 + 63) / 64, BT_ / 128);    // [BT x 4D] GEMM
  dim3 gV((VV + 63) / 64, BT_ / 128);    // LM head
  dim3 gA(TT / 64, HH, BB);              // attention

  k_embed<<<BT_, 256, 0, stream>>>(idx, tok, pos, xA);

  for (int l = 0; l < LL; ++l) {
    const size_t wOff  = (size_t)l * DD * DD;
    const size_t w1Off = (size_t)l * DD * D4;
    const size_t w2Off = (size_t)l * D4 * DD;

    k_ln<<<BT_, 256, 0, stream>>>(xA, ln1_s + (size_t)l * DD, ln1_b + (size_t)l * DD, xn);

    k_gemm<<<gD, 128, 0, stream>>>(xn, Wq + wOff, nullptr, nullptr, nullptr, q,
                                   BT_, DD, DD, 0);
    k_gemm<<<gD, 128, 0, stream>>>(xn, Wk + wOff, nullptr, nullptr, nullptr, k,
                                   BT_, DD, DD, 0);
    k_gemm<<<gD, 128, 0, stream>>>(xn, Wv + wOff, nullptr, nullptr, nullptr, v,
                                   BT_, DD, DD, 0);

    k_attn<<<gA, 128, 0, stream>>>(q, k, v, attn);

    // x = x + attn @ Wo + bo   (xA -> xB)
    k_gemm<<<gD, 128, 0, stream>>>(attn, Wo + wOff, bo + (size_t)l * DD, xA, xB, nullptr,
                                   BT_, DD, DD, 0);

    k_ln<<<BT_, 256, 0, stream>>>(xB, ln2_s + (size_t)l * DD, ln2_b + (size_t)l * DD, xn);

    // h = relu(xn @ W1 + b1)
    k_gemm<<<g4, 128, 0, stream>>>(xn, W1 + w1Off, b1 + (size_t)l * D4, nullptr, nullptr,
                                   hbuf, BT_, D4, DD, 1);
    // x = x + h @ W2 + b2      (xB -> xA)
    k_gemm<<<gD, 128, 0, stream>>>(hbuf, W2 + w2Off, b2 + (size_t)l * DD, xB, xA, nullptr,
                                   BT_, DD, D4, 0);
  }

  k_ln<<<BT_, 256, 0, stream>>>(xA, lnf_s, lnf_b, xn);

  // logits = xn @ lm_W + lm_b
  k_gemm<<<gV, 128, 0, stream>>>(xn, lm_W, lm_b, nullptr, logits, nullptr,
                                 BT_, VV, DD, 0);
}
